// LSM_24326694765087
// MI455X (gfx1250) — matrix operand order repaired and checked by reading, not compiled, not run on hardware
//
#include <hip/hip_runtime.h>
#include <hip/hip_bf16.h>
#include <stdint.h>

// ---------------- problem constants ----------------
#define INP   1024
#define NN    8000
#define NE    6400
#define NI    1600
#define AA    (INP + NN)          // 9024 presynaptic rows (= 282*32, exact)
#define TS    1000
#define PRB0  200
#define PRB1  1000
#define DT    0.001f
#define TAU   0.02f
#define DECAY (1.0f - DT / TAU)   // 0.95
#define VTH   0.5f
#define RATE  0.5f
#define NORM  800.0f

#define NKT      282              // K tiles of 32 along A dimension
#define NTE      400              // NE/16 column tiles (E)
#define NTI      100              // NI/16 column tiles (I)
#define NT_TOT   (NTE + NTI)      // 500
#define KCH      6                // K-chunks per column tile (282 = 6*47)
#define KT_PER_CH 47
#define MV_WAVES (NT_TOT * KCH)   // 3000 waves
#define MV_BLOCKS (MV_WAVES / 8)  // 375 blocks of 8 waves

typedef __attribute__((ext_vector_type(16))) __bf16 v16bf;
typedef __attribute__((ext_vector_type(8)))  __bf16 v8bf;
typedef __attribute__((ext_vector_type(8)))  float  v8f;

__device__ __forceinline__ __bf16 f2bf(float f) {
    unsigned u = __float_as_uint(f);
    unsigned r = (u + 0x7FFFu + ((u >> 16) & 1u)) >> 16;   // round-to-nearest-even
    unsigned short s = (unsigned short)r;
    __bf16 b;
    __builtin_memcpy(&b, &s, 2);
    return b;
}

// -------- one-time per call: pack fp32 weights into bf16 WMMA-B tile layout ----
// Tile (nt,kt) stored at element offset (nt*nkt + kt)*512; within a tile,
// element [lane*16 + j] = W[kt*32 + 16*(lane>>4) + j][nt*16 + (lane&15)]
// which is exactly the v16bf B operand each lane feeds to v_wmma_*_bf16.
__global__ void lsm_pack_w(const float* __restrict__ W, __bf16* __restrict__ out,
                           int ncols, int nkt, long total) {
    long gid = (long)blockIdx.x * blockDim.x + threadIdx.x;
    if (gid >= total) return;
    int  tile = (int)(gid >> 9);
    int  r    = (int)(gid & 511);
    int  col  = r & 15;          // fastest-varying -> coalesced source reads
    int  kk   = r >> 4;          // 0..31
    int  half = kk >> 4;
    int  j    = kk & 15;
    int  lane = (half << 4) | col;
    int  nt   = tile / nkt;
    int  kt   = tile - nt * nkt;
    long k    = (long)kt * 32 + kk;
    long n    = (long)nt * 16 + col;
    float v   = W[k * (long)ncols + n];
    out[((long)tile << 9) + lane * 16 + j] = f2bf(v);
}

// -------- one-time: Poisson input spikes for all TS steps, as bf16 {0,1} ------
__global__ void lsm_spike_table(const float* __restrict__ rand_p,
                                const float* __restrict__ inp,
                                __bf16* __restrict__ sp, long total) {
    long gid = (long)blockIdx.x * blockDim.x + threadIdx.x;
    if (gid >= total) return;
    int i = (int)(gid % INP);
    sp[gid] = f2bf((rand_p[gid] <= inp[i] * RATE) ? 1.0f : 0.0f);
}

// -------- one-time: state init (vmem/acc/cnt zero, t=0 input vectors) ---------
__global__ void lsm_init_state(float* vmem, float* acc, float* cnt,
                               __bf16* in_e, __bf16* in_i,
                               const __bf16* __restrict__ sp) {
    int i = blockIdx.x * blockDim.x + threadIdx.x;
    if (i < AA) {
        __bf16 v = (i < INP) ? sp[i] : f2bf(0.0f);
        in_e[i] = v;
        in_i[i] = v;
    }
    if (i < NN) { vmem[i] = 0.0f; acc[i] = 0.0f; }
    if (i < NE) cnt[i] = 0.0f;
}

// -------- per step, kernel 1: WMMA matvec with broadcast-A spike vector -------
__global__ void __launch_bounds__(256)
lsm_matvec(const __bf16* __restrict__ wpe, const __bf16* __restrict__ wpi,
           const __bf16* __restrict__ in_e, const __bf16* __restrict__ in_i,
           float* __restrict__ acc) {
    __shared__ __align__(16) __bf16 s_in[2][AA];

    // cooperative load of both 9024-element bf16 spike vectors into LDS (b128)
    {
        const uint4* ge = (const uint4*)in_e;   // 9024 bf16 = 1128 uint4
        const uint4* gi = (const uint4*)in_i;
        uint4* se = (uint4*)&s_in[0][0];
        uint4* si = (uint4*)&s_in[1][0];
        for (int i = threadIdx.x; i < AA / 8; i += 256) {
            se[i] = ge[i];
            si[i] = gi[i];
        }
    }
    __syncthreads();

    const int wave = blockIdx.x * 8 + (threadIdx.x >> 5);  // wave32 (gfx1250)
    const int lane = threadIdx.x & 31;
    const int ntile = wave / KCH;
    const int kch   = wave - ntile * KCH;

    const bool isE = ntile < NTE;
    const int  ct  = isE ? ntile : (ntile - NTE);
    const int  kt0 = kch * KT_PER_CH;

    // Single byte base for the whole K-chunk: every unrolled load below becomes
    // global_load_b128 [base], offset:<imm>  (chunk spans 47 KB << 24-bit imm).
    const char* __restrict__ wb =
        (const char*)((isE ? wpe : wpi) + (long)ct * NKT * 512 + lane * 16) +
        (size_t)kt0 * 1024;
    const __bf16* __restrict__ sv =
        (isE ? &s_in[0][0] : &s_in[1][0]) + kt0 * 32 + (lane >> 4) * 16;
    float* __restrict__ accp = acc + (isE ? ct * 16 : NE + ct * 16);

    // Two accumulators to halve the WMMA C->C dependency chain.
    v8f c0 = {}, c1 = {};
    #pragma unroll
    for (int i = 0; i < KT_PER_CH; ++i) {
        const v8bf blo = *(const v8bf*)(wb + (size_t)i * 1024);
        const v8bf bhi = *(const v8bf*)(wb + (size_t)i * 1024 + 16);
        v16bf b = __builtin_shufflevector(blo, bhi,
                   0,1,2,3,4,5,6,7,8,9,10,11,12,13,14,15);
        const v8bf alo = *(const v8bf*)(sv + i * 32);
        const v8bf ahi = *(const v8bf*)(sv + i * 32 + 8);
        v16bf a = __builtin_shufflevector(alo, ahi,
                   0,1,2,3,4,5,6,7,8,9,10,11,12,13,14,15);
        if (i & 1)
            c1 = __builtin_amdgcn_wmma_f32_16x16x32_bf16(
                     false, a, false, b, (short)0, c1, false, false);
        else
            c0 = __builtin_amdgcn_wmma_f32_16x16x32_bf16(
                     false, a, false, b, (short)0, c0, false, false);
    }
    v8f c = c0 + c1;
    // All 16 D rows are identical; row M=0 lives in c[0] of lanes 0..15.
    if (lane < 16) atomicAdd(&accp[lane], c[0]);
}

// -------- per step, kernel 2: LIF update + history rotation + probe ----------
__global__ void lsm_update(float* __restrict__ vmem, float* __restrict__ acc,
                           float* __restrict__ cnt,
                           __bf16* __restrict__ in_e, __bf16* __restrict__ in_i,
                           const __bf16* __restrict__ sp, int t) {
    int n = blockIdx.x * blockDim.x + threadIdx.x;
    if (n < INP && t + 1 < TS) {
        __bf16 s = sp[(long)(t + 1) * INP + n];   // next step's input spikes
        in_e[n] = s;
        in_i[n] = s;
    }
    if (n < NN) {
        float v = vmem[n] * DECAY + acc[n];
        float s = (v >= VTH) ? 1.0f : 0.0f;
        vmem[n] = v * (1.0f - s);
        acc[n]  = 0.0f;                            // ready for next matvec
        if (n < NE && t >= PRB0 && t < PRB1) cnt[n] += s;
        // E part: in_e holds slp, in_i holds sl;  I part: swapped.
        int p = INP + n;
        if (n < NE) {
            __bf16 old_sl = in_i[p];
            in_e[p] = old_sl;      // new slp = old sl
            in_i[p] = f2bf(s);     // new sl
        } else {
            __bf16 old_sl = in_e[p];
            in_i[p] = old_sl;
            in_e[p] = f2bf(s);
        }
    }
}

__global__ void lsm_finalize(const float* __restrict__ cnt, float* __restrict__ out) {
    int n = blockIdx.x * blockDim.x + threadIdx.x;
    if (n < NE) out[n] = cnt[n] * (1.0f / NORM);
}

extern "C" void kernel_launch(void* const* d_in, const int* in_sizes, int n_in,
                              void* d_out, int out_size, void* d_ws, size_t ws_size,
                              hipStream_t stream) {
    const float* inp    = (const float*)d_in[0];   // [1024]
    const float* w_ae   = (const float*)d_in[1];   // [9024*6400]
    const float* w_ai   = (const float*)d_in[2];   // [9024*1600]
    const float* rand_p = (const float*)d_in[3];   // [1000*1024]
    float* out = (float*)d_out;

    // ---- workspace carve-up (aligned to 256B) ----
    char* ws = (char*)d_ws;
    size_t off = 0;
    auto carve = [&](size_t bytes) {
        char* p = ws + off;
        off = (off + bytes + 255) & ~(size_t)255;
        return p;
    };
    const long wpe_elems = (long)NTE * NKT * 512;  // 57,753,600
    const long wpi_elems = (long)NTI * NKT * 512;  // 14,438,400
    __bf16* wpe  = (__bf16*)carve(wpe_elems * 2);  // ~115.5 MB
    __bf16* wpi  = (__bf16*)carve(wpi_elems * 2);  // ~28.9 MB
    __bf16* spt  = (__bf16*)carve((long)TS * INP * 2);
    __bf16* in_e = (__bf16*)carve(AA * 2);
    __bf16* in_i = (__bf16*)carve(AA * 2);
    float*  vmem = (float*)carve(NN * 4);
    float*  acc  = (float*)carve(NN * 4);
    float*  cnt  = (float*)carve(NE * 4);
    (void)ws_size;

    // ---- one-time (per call) setup: pack weights to bf16 tiles, spike table ----
    lsm_pack_w<<<(unsigned)((wpe_elems + 255) / 256), 256, 0, stream>>>(
        w_ae, wpe, NE, NKT, wpe_elems);
    lsm_pack_w<<<(unsigned)((wpi_elems + 255) / 256), 256, 0, stream>>>(
        w_ai, wpi, NI, NKT, wpi_elems);
    lsm_spike_table<<<(TS * INP + 255) / 256, 256, 0, stream>>>(
        rand_p, inp, spt, (long)TS * INP);
    lsm_init_state<<<(AA + 255) / 256, 256, 0, stream>>>(
        vmem, acc, cnt, in_e, in_i, spt);

    // ---- 1000 sequential steps: matvec (WMMA) then LIF update ----
    for (int t = 0; t < TS; ++t) {
        lsm_matvec<<<MV_BLOCKS, 256, 0, stream>>>(wpe, wpi, in_e, in_i, acc);
        lsm_update<<<(NN + 255) / 256, 256, 0, stream>>>(
            vmem, acc, cnt, in_e, in_i, spt, t);
    }
    lsm_finalize<<<(NE + 255) / 256, 256, 0, stream>>>(cnt, out);
}
